// RNNDecoder_4475355922744
// MI455X (gfx1250) — matrix operand order repaired and checked by reading, not compiled, not run on hardware
//
#include <hip/hip_runtime.h>

// ---------------------------------------------------------------------------
// Problem constants (B=16, T=128, V=32000, E=256, H=512)
// ---------------------------------------------------------------------------
#define BB 16
#define TT 128
#define VV 32000
#define EE 256
#define HH 512
#define BT (BB * TT)          // 2048
#define SOS_TOKEN 1

typedef __attribute__((ext_vector_type(16))) __bf16 v16bf;
typedef __attribute__((ext_vector_type(8)))  float  v8f;
typedef __attribute__((ext_vector_type(2)))  float  v2f;

#if __has_builtin(__builtin_amdgcn_wmma_f32_16x16x4_f32)
#define USE_WMMA_F32 1
#endif

__device__ __forceinline__ unsigned short f2bf_u16(float f) {
    unsigned u = __builtin_bit_cast(unsigned, f);
    // round-to-nearest-even bf16 truncation
    return (unsigned short)((u + 0x7FFFu + ((u >> 16) & 1u)) >> 16);
}

// gfx1250 async copy: global -> LDS, 16 bytes per lane, tracked by ASYNCcnt.
__device__ __forceinline__ void async_copy_b128(unsigned lds_off, const void* gptr) {
    asm volatile("global_load_async_to_lds_b128 %0, %1, off"
                 :: "v"(lds_off), "v"(gptr) : "memory");
}
template <int N>
__device__ __forceinline__ void wait_asynccnt() {
#if __has_builtin(__builtin_amdgcn_s_wait_asynccnt)
    __builtin_amdgcn_s_wait_asynccnt(N);
#else
    asm volatile("s_wait_asynccnt %0" :: "i"(N) : "memory");
#endif
}

// ---------------------------------------------------------------------------
// Kernel 1: W_out (f32 [V,H]) -> bf16 workspace copy
// ---------------------------------------------------------------------------
__global__ __launch_bounds__(256) void k_cvt_bf16(const float* __restrict__ src,
                                                  unsigned short* __restrict__ dst,
                                                  long n4) {
    long i = (long)blockIdx.x * 256 + threadIdx.x;
    if (i >= n4) return;
    float4 f = ((const float4*)src)[i];
    ushort4 o;
    o.x = f2bf_u16(f.x); o.y = f2bf_u16(f.y);
    o.z = f2bf_u16(f.z); o.w = f2bf_u16(f.w);
    ((ushort4*)dst)[i] = o;
}

// ---------------------------------------------------------------------------
// Kernel 2: embedding gather + input projection
//   x_proj[bt][n] = sum_e emb[token(bt)][e] * W_ih[n][e] + b_ih[n]
// ---------------------------------------------------------------------------
__global__ __launch_bounds__(256) void k_embed_xproj(const int* __restrict__ target,
                                                     const float* __restrict__ embedding,
                                                     const float* __restrict__ W_ih,
                                                     const float* __restrict__ b_ih,
                                                     float* __restrict__ xproj) {
    const int bt = blockIdx.x;                 // 0..2047 = b*128 + t
    const int b = bt >> 7, t = bt & 127;
    const int tok = (t == 0) ? SOS_TOKEN : target[b * TT + t - 1];
    __shared__ float4 semb[EE / 4];            // 256 floats
    if (threadIdx.x < EE / 4)
        semb[threadIdx.x] = ((const float4*)(embedding + (long)tok * EE))[threadIdx.x];
    __syncthreads();
    for (int n = threadIdx.x; n < HH; n += 256) {
        const float4* wr = (const float4*)(W_ih + (long)n * EE);
        float acc = 0.f;
#pragma unroll 8
        for (int i = 0; i < EE / 4; ++i) {
            float4 e = semb[i], w = wr[i];
            acc += e.x * w.x + e.y * w.y + e.z * w.z + e.w * w.w;
        }
        xproj[(long)bt * HH + n] = acc + b_ih[n];
    }
}

// ---------------------------------------------------------------------------
// Kernel 3: tanh-RNN recurrence, single workgroup (serial over T).
//   h_t = tanh(x_proj_t + h_{t-1} @ W_hh^T + b_hh)
// h in LDS, stride 516 floats (16 rows -> 16 distinct banks). 16 waves;
// wave w owns output columns [32w, 32w+32): two 16x16 tiles per step via
// v_wmma_f32_16x16x4_f32. Emits bf16 hs (for the big GEMM) and f32 h_last.
// ---------------------------------------------------------------------------
#define HSTRIDE 516
__global__ __launch_bounds__(512) void k_recurrence(const float* __restrict__ enc_h,
                                                    const float* __restrict__ W_hh,
                                                    const float* __restrict__ b_hh,
                                                    const float* __restrict__ xproj,
                                                    unsigned short* __restrict__ hs_bf,
                                                    float* __restrict__ h_last) {
    __shared__ float h[BB * HSTRIDE];          // ~33 KB
    const int tid = threadIdx.x;
    for (int i = tid; i < BB * HH; i += 512)
        h[(i >> 9) * HSTRIDE + (i & 511)] = enc_h[i];
    __syncthreads();

#ifdef USE_WMMA_F32
    const int wave = tid >> 5;
    const int lane = tid & 31;
    const int lane16 = lane & 15;
    const int half = lane >> 4;                // 0: lanes 0-15, 1: lanes 16-31
    const int koff = half * 2;                 // A/B K sub-offset per half-wave
    const int n0 = wave * 32;                  // this wave's column base

    for (int t = 0; t < TT; ++t) {
        v8f acc0 = {}, acc1 = {};
        const float* hrow = &h[lane16 * HSTRIDE];
        const float* w0row = &W_hh[(long)(n0 + lane16) * HH];
        const float* w1row = &W_hh[(long)(n0 + 16 + lane16) * HH];
        for (int k0 = 0; k0 < HH; k0 += 4) {
            float2 af = *(const float2*)(hrow + k0 + koff);
            float2 bf0 = *(const float2*)(w0row + k0 + koff);
            float2 bf1 = *(const float2*)(w1row + k0 + koff);
            v2f a;  a.x = af.x;  a.y = af.y;
            v2f b0; b0.x = bf0.x; b0.y = bf0.y;
            v2f b1; b1.x = bf1.x; b1.y = bf1.y;
            acc0 = __builtin_amdgcn_wmma_f32_16x16x4_f32(false, a, false, b0,
                                                         (short)0, acc0, false, false);
            acc1 = __builtin_amdgcn_wmma_f32_16x16x4_f32(false, a, false, b1,
                                                         (short)0, acc1, false, false);
        }
        __syncthreads();                       // everyone done reading h[t-1]
#pragma unroll
        for (int r = 0; r < 8; ++r) {
            const int m = r + half * 8;        // batch row in [0,16)
            {
                const int n = n0 + lane16;
                const long bt = (long)m * TT + t;
                float v = acc0[r] + xproj[bt * HH + n] + b_hh[n];
                float hn = tanhf(v);
                h[m * HSTRIDE + n] = hn;
                hs_bf[bt * HH + n] = f2bf_u16(hn);
                if (t == TT - 1) h_last[m * HH + n] = hn;
            }
            {
                const int n = n0 + 16 + lane16;
                const long bt = (long)m * TT + t;
                float v = acc1[r] + xproj[bt * HH + n] + b_hh[n];
                float hn = tanhf(v);
                h[m * HSTRIDE + n] = hn;
                hs_bf[bt * HH + n] = f2bf_u16(hn);
                if (t == TT - 1) h_last[m * HH + n] = hn;
            }
        }
        __syncthreads();                       // h[t] fully written
    }
#else
    const int n = tid;
    for (int t = 0; t < TT; ++t) {
        float acc[BB];
#pragma unroll
        for (int b = 0; b < BB; ++b) acc[b] = 0.f;
        const float* wrow = &W_hh[(long)n * HH];
        for (int k = 0; k < HH; ++k) {
            float w = wrow[k];
#pragma unroll
            for (int b = 0; b < BB; ++b) acc[b] += h[b * HSTRIDE + k] * w;
        }
        __syncthreads();
#pragma unroll
        for (int b = 0; b < BB; ++b) {
            const long bt = (long)b * TT + t;
            float hn = tanhf(acc[b] + xproj[bt * HH + n] + b_hh[n]);
            h[b * HSTRIDE + n] = hn;
            hs_bf[bt * HH + n] = f2bf_u16(hn);
            if (t == TT - 1) h_last[b * HH + n] = hn;
        }
        __syncthreads();
    }
#endif
}

// ---------------------------------------------------------------------------
// Kernel 4: logits GEMM  hs[2048x512](bf16) @ W_out^T[512x32000](bf16), f32 acc.
// Block = 8 waves = 64 M-rows x 128 N-cols. Each wave: 1 N-tile x 4 M-tiles
// (B fragment reused 4x -> W_out L2 traffic cut 4x vs 16-row M blocks).
// The shared 64x32 bf16 A panel per K-step is staged in LDS once per block
// with global_load_async_to_lds_b128, double-buffered on ASYNCcnt.
// LDS rows padded to 80 B so 16-lane fragment reads hit 16 distinct banks.
//
// Fragment layouts (CDNA5 ISA 7.12.2):
//   A (16x32 bf16): lane<16 -> M=lane, elems = K[0..7] then K[16..23]
//                   lane>=16 -> same M, K[8..15] then K[24..31]
//   B (32x16 bf16): lane<16 -> N=lane, K[0..7] then K[16..23]; lane>=16 -> same N, K[8..15] then K[24..31]
//   C/D (f32): VGPR r, lane<16 -> (M=r, N=lane); lane>=16 -> (M=r+8, N=lane-16)
// ---------------------------------------------------------------------------
#define AROW 80                                 // padded LDS row pitch (bytes)
#define ABUF (64 * AROW)                        // 5120 B per buffer

union Frag16 { v16bf v; uint4 q[2]; };

__global__ __launch_bounds__(256) void k_logits(const unsigned short* __restrict__ hs_bf,
                                                const unsigned short* __restrict__ wout_bf,
                                                const float* __restrict__ b_out,
                                                float* __restrict__ out) {
    __shared__ __attribute__((aligned(16))) unsigned char abuf[2 * ABUF];
    const int tid = threadIdx.x;
    const int wave = tid >> 5;
    const int lane = tid & 31;
    const int lane16 = lane & 15;
    const int half = lane >> 4;

    const int m0 = blockIdx.x * 64;                   // 64-row M panel
    const int colBase = blockIdx.y * 128 + wave * 16; // this wave's N tile

    // staging assignment: thread -> (row, 16B segment) of the 64x32 bf16 panel
    const int srow = tid >> 2;                        // 0..63
    const int sseg = tid & 3;                         // 0..3 (4 x 16B per row)
    const unsigned short* aG = hs_bf + (long)(m0 + srow) * HH + sseg * 8;
    const unsigned ldsBase = (unsigned)(unsigned long long)&abuf[0];
    const unsigned ldsMine = ldsBase + (unsigned)(srow * AROW + sseg * 16);

    v8f acc[4] = {{}, {}, {}, {}};

    // prologue: stage K-step 0 into buffer 0
    async_copy_b128(ldsMine, aG);

    const unsigned short* bRow = wout_bf + (long)(colBase + lane16) * HH + half * 8;

    for (int ks = 0; ks < HH / 32; ++ks) {            // 16 K-steps
        const int k0 = ks * 32;
        const int cur = ks & 1;
        if (ks + 1 < HH / 32) {                        // prefetch next panel
            async_copy_b128(ldsMine + ((ks + 1) & 1) * ABUF, aG + (k0 + 32));
            wait_asynccnt<1>();                        // oldest copy complete
        } else {
            wait_asynccnt<0>();                        // last panel complete
        }
        __syncthreads();                               // panel visible to all waves

        Frag16 Bf;                                     // this wave's B fragment
        Bf.q[0] = *(const uint4*)(bRow + k0);
        Bf.q[1] = *(const uint4*)(bRow + k0 + 16);

        const unsigned char* lp = &abuf[cur * ABUF];
#pragma unroll
        for (int mt = 0; mt < 4; ++mt) {
            const unsigned char* ap = lp + (mt * 16 + lane16) * AROW + half * 16;
            Frag16 A;
            A.q[0] = *(const uint4*)(ap);              // K[0..7]  (or 8..15)
            A.q[1] = *(const uint4*)(ap + 32);         // K[16..23](or 24..31)
            acc[mt] = __builtin_amdgcn_wmma_f32_16x16x32_bf16(
                false, A.v, false, Bf.v, (short)0, acc[mt], false, false);
        }
        __syncthreads();                               // done reading cur buffer
    }

    const int col = colBase + lane16;
    const float bo = b_out[col];
#pragma unroll
    for (int mt = 0; mt < 4; ++mt) {
#pragma unroll
        for (int r = 0; r < 8; ++r) {
            const int m = m0 + mt * 16 + r + half * 8;
            out[(long)m * VV + col] = acc[mt][r] + bo;
        }
    }
}

// ---------------------------------------------------------------------------
// Kernel 5: in-place log_softmax over V=32000 per row; one block per row.
// ---------------------------------------------------------------------------
__global__ __launch_bounds__(256) void k_logsoftmax(float* __restrict__ out) {
    const long row = blockIdx.x;
    float4* p4 = (float4*)(out + row * (long)VV);
    const int tid = threadIdx.x;
    __shared__ float red[256];

    float mx = -3.4e38f;
    for (int i = tid; i < VV / 4; i += 256) {
        float4 v = p4[i];
        mx = fmaxf(mx, fmaxf(fmaxf(v.x, v.y), fmaxf(v.z, v.w)));
    }
    red[tid] = mx; __syncthreads();
    for (int s = 128; s > 0; s >>= 1) {
        if (tid < s) red[tid] = fmaxf(red[tid], red[tid + s]);
        __syncthreads();
    }
    mx = red[0]; __syncthreads();

    float sum = 0.f;
    for (int i = tid; i < VV / 4; i += 256) {
        float4 v = p4[i];
        sum += __expf(v.x - mx) + __expf(v.y - mx) + __expf(v.z - mx) + __expf(v.w - mx);
    }
    red[tid] = sum; __syncthreads();
    for (int s = 128; s > 0; s >>= 1) {
        if (tid < s) red[tid] += red[tid + s];
        __syncthreads();
    }
    const float lse = mx + __logf(red[0]);
    __syncthreads();

    for (int i = tid; i < VV / 4; i += 256) {
        float4 v = p4[i];
        v.x -= lse; v.y -= lse; v.z -= lse; v.w -= lse;
        p4[i] = v;
    }
}

// ---------------------------------------------------------------------------
// Launch. Inputs (setup_inputs order):
//  0 encoder_outputs (unused)  1 encoder_hidden  2 target_tensor  3 embedding
//  4 W_ih  5 W_hh  6 b_ih  7 b_hh  8 W_out  9 b_out
// d_out: log_probs [16,128,32000] f32 (65,536,000) ++ h_last [1,16,512] (8,192)
// Workspace: [W_out bf16: 32,768,000 B][x_proj f32: 4,194,304 B][hs bf16: 2,097,152 B]
// ---------------------------------------------------------------------------
extern "C" void kernel_launch(void* const* d_in, const int* in_sizes, int n_in,
                              void* d_out, int out_size, void* d_ws, size_t ws_size,
                              hipStream_t stream) {
    const float* enc_h     = (const float*)d_in[1];
    const int*   target    = (const int*)d_in[2];
    const float* embedding = (const float*)d_in[3];
    const float* W_ih      = (const float*)d_in[4];
    const float* W_hh      = (const float*)d_in[5];
    const float* b_ih      = (const float*)d_in[6];
    const float* b_hh      = (const float*)d_in[7];
    const float* W_out     = (const float*)d_in[8];
    const float* b_out     = (const float*)d_in[9];
    float* out = (float*)d_out;

    char* ws = (char*)d_ws;
    unsigned short* wout_bf = (unsigned short*)ws;                       // 32,768,000 B
    float*          xproj   = (float*)(ws + 32768000);                   //  4,194,304 B
    unsigned short* hs_bf   = (unsigned short*)(ws + 32768000 + 4194304);//  2,097,152 B

    // 1) W_out -> bf16
    k_cvt_bf16<<<(VV * HH / 4 + 255) / 256, 256, 0, stream>>>(W_out, wout_bf,
                                                              (long)VV * HH / 4);
    // 2) embedding gather + input projection
    k_embed_xproj<<<BT, 256, 0, stream>>>(target, embedding, W_ih, b_ih, xproj);
    // 3) serial recurrence (fp32 WMMA), emits bf16 hs + f32 h_last
    k_recurrence<<<1, 512, 0, stream>>>(enc_h, W_hh, b_hh, xproj, hs_bf,
                                        out + (long)BT * VV);
    // 4) logits GEMM (bf16 WMMA + async-LDS A staging) -> d_out
    k_logits<<<dim3(BT / 64, VV / 128), 256, 0, stream>>>(hs_bf, wout_bf, b_out, out);
    // 5) in-place log_softmax
    k_logsoftmax<<<BT, 256, 0, stream>>>(out);
}
